// IMDecoderLayer_27324581937563
// MI455X (gfx1250) — matrix operand order repaired and checked by reading, not compiled, not run on hardware
//
#include <hip/hip_runtime.h>
#include <hip/hip_bf16.h>
#include <cstdint>
#include <cstddef>

// ---------------------------------------------------------------------------
// Fused IMDecoderLayer for MI455X (gfx1250, wave32, WMMA).
//
// Pipeline (all on `stream`, workspace layout below):
//   prep    : hidden fp32 -> f16 A-matrix; zero out-accumulator; init rowmax keys
//   gemm1   : logits = A @ W_e2v^T  (f16 WMMA, f32 acc). Stores f16 logits to ws,
//             folds per-row max via monotone-uint atomicMax.
//   softneg : in-place  p = (len>0) ? -exp(logit - rowmax) : 0   (f16, 8-wide)
//   scatter : p[m, mask_idx[m,k]] = 1/len  for k < len   (<=4096 writes)
//   gemm2   : acc += P @ W_v2e^T   (f16 WMMA, split-K over V, fp32 atomicAdd)
//   final   : out = acc*strength + bias + hidden
//
// Workspace requirement: ~151 MiB
//   [0)          A16   512*896   f16      =    917504 B
//   [917504)     key   512       u32      =      2048 B
//   [919552)     acc   512*896   f32      =   1835008 B
//   [2754560)    P     512*V     f16      = 155582464 B
// ---------------------------------------------------------------------------

typedef __attribute__((ext_vector_type(16))) _Float16 v16h;
typedef __attribute__((ext_vector_type(8)))  _Float16 v8h;
typedef __attribute__((ext_vector_type(4)))  _Float16 v4h;
typedef __attribute__((ext_vector_type(8)))  float    v8f;

#define MDIM 512        // B*S
#define HDIM 896
#define VDIM 151936     // = 64 * 2374, = 32 * 4748
#define KTOK 8
#define BP   40         // LDS B-tile pitch in halves (16B-aligned, conflict pad)
#define G2Y  32         // split-K chunks in gemm2
#define KST2 (VDIM/32)  // 4748 K-steps in gemm2
#define SPW  ((KST2 + G2Y - 1) / G2Y) // 149 steps per chunk

// monotone float<->uint order encoding (for atomicMax on f32 row maxima)
__device__ __forceinline__ unsigned enc_f32(float f) {
    unsigned u = __float_as_uint(f);
    return (u & 0x80000000u) ? ~u : (u | 0x80000000u);
}
__device__ __forceinline__ float dec_f32(unsigned k) {
    unsigned u = (k & 0x80000000u) ? (k ^ 0x80000000u) : ~k;
    return __uint_as_float(u);
}

__global__ void imdl_prep(const float* __restrict__ hidden,
                          _Float16* __restrict__ A16,
                          float* __restrict__ acc,
                          unsigned* __restrict__ key) {
    int i = blockIdx.x * blockDim.x + threadIdx.x;
    if (i < MDIM * HDIM) { A16[i] = (_Float16)hidden[i]; acc[i] = 0.0f; }
    if (i < MDIM) key[i] = 0x007FFFFFu;   // enc(-inf)
}

// GEMM1: 16 waves own full M=512 (32 rows each); block owns 64 columns of V.
// W_e2v streamed from HBM exactly once, converted fp32->f16 into LDS.
__global__ __launch_bounds__(512)
void imdl_gemm1(const _Float16* __restrict__ A,   // [512, 896] f16
                const float* __restrict__ W,      // [V, 896] fp32
                _Float16* __restrict__ P,         // [512, V] f16 logits out
                unsigned* __restrict__ key) {
    __shared__ _Float16 Bs[64 * BP];
    const int tid   = threadIdx.x;
    const int wave  = tid >> 5;
    const int lane  = tid & 31;
    const int lo    = lane & 15;
    const int hi    = lane >> 4;
    const int nbase = blockIdx.x * 64;
    const int mbase = wave * 32;

    v8f acc[2][4] = {};

    for (int kc = 0; kc < HDIM; kc += 32) {
        // cooperative stage: W[nbase..+64, kc..+32] fp32 -> f16 LDS
        {
            const int row = tid >> 3;        // 0..63
            const int col = (tid & 7) * 4;   // 0..28
            const float4 w4 = *(const float4*)(W + (size_t)(nbase + row) * HDIM + kc + col);
            v4h h4;
            h4[0] = (_Float16)w4.x; h4[1] = (_Float16)w4.y;
            h4[2] = (_Float16)w4.z; h4[3] = (_Float16)w4.w;
            *(v4h*)(&Bs[row * BP + col]) = h4;
        }
        __syncthreads();

        // A operands: lane = row m, K chunks [hi*8, hi*8+8) and [16+hi*8, ...)
        v16h a[2];
#pragma unroll
        for (int mt = 0; mt < 2; ++mt) {
            const _Float16* ar = A + (size_t)(mbase + mt * 16 + lo) * HDIM + kc;
            v8h a0 = *(const v8h*)(ar + hi * 8);
            v8h a1 = *(const v8h*)(ar + 16 + hi * 8);
            a[mt] = __builtin_shufflevector(a0, a1, 0,1,2,3,4,5,6,7,8,9,10,11,12,13,14,15);
        }
#pragma unroll
        for (int nt = 0; nt < 4; ++nt) {
            // B operand: lane = column n, K chunks [hi*8, hi*8+8) and [16+hi*8, ...)
            const _Float16* br = &Bs[(nt * 16 + lo) * BP + hi * 8];
            v8h b0 = *(const v8h*)br;
            v8h b1 = *(const v8h*)(br + 16);
            v16h b = __builtin_shufflevector(b0, b1, 0,1,2,3,4,5,6,7,8,9,10,11,12,13,14,15);
#pragma unroll
            for (int mt = 0; mt < 2; ++mt)
                acc[mt][nt] = __builtin_amdgcn_wmma_f32_16x16x32_f16(
                    false, a[mt], false, b, (short)0, acc[mt][nt], false, false);
        }
        __syncthreads();
    }

    // epilogue: f16 logit store + fold row-max (C/D layout: m = r + hi*8, n = lo)
#pragma unroll
    for (int mt = 0; mt < 2; ++mt) {
#pragma unroll
        for (int r = 0; r < 8; ++r) {
            const int m = mbase + mt * 16 + hi * 8 + r;
            float mx = -3.402823466e38f;
#pragma unroll
            for (int nt = 0; nt < 4; ++nt) {
                const float v = acc[mt][nt][r];
                mx = fmaxf(mx, v);
                P[(size_t)m * VDIM + nbase + nt * 16 + lo] = (_Float16)v;
            }
            mx = fmaxf(mx, __shfl_xor(mx, 1, 32));
            mx = fmaxf(mx, __shfl_xor(mx, 2, 32));
            mx = fmaxf(mx, __shfl_xor(mx, 4, 32));
            mx = fmaxf(mx, __shfl_xor(mx, 8, 32));
            if (lo == 0) atomicMax(&key[m], enc_f32(mx));
        }
    }
}

// elementwise: p = (len>0) ? -exp(l - rowmax) : 0, 8 halves / thread
__global__ void imdl_softneg(_Float16* __restrict__ P,
                             const unsigned* __restrict__ key,
                             const int* __restrict__ mask_len) {
    const int CH = VDIM / 8;     // 18992 chunks per row
    int t = blockIdx.x * blockDim.x + threadIdx.x;
    if (t >= MDIM * CH) return;
    const int m = t / CH;
    const int c = (t - m * CH) * 8;
    _Float16* p = P + (size_t)m * VDIM + c;
    v8h x = *(const v8h*)p;
    const int len = mask_len[m];
    if (len > 0) {
        const float mx = dec_f32(key[m]);
#pragma unroll
        for (int i = 0; i < 8; ++i)
            x[i] = (_Float16)(-__expf((float)x[i] - mx));
    } else {
#pragma unroll
        for (int i = 0; i < 8; ++i) x[i] = (_Float16)0.0f;
    }
    *(v8h*)p = x;
}

// ragged scatter: p[m, idx[m,k]] = 1/len for k < len (duplicates write same value)
__global__ void imdl_scatter(_Float16* __restrict__ P,
                             const int* __restrict__ mask_idx,
                             const int* __restrict__ mask_len) {
    int t = blockIdx.x * blockDim.x + threadIdx.x;
    if (t >= MDIM * KTOK) return;
    const int m = t >> 3;
    const int k = t & 7;
    const int len = mask_len[m];
    if (k < len) {
        const int col = mask_idx[t];
        P[(size_t)m * VDIM + col] = (_Float16)(1.0f / (float)len);
    }
}

// GEMM2: block = (H-tile of 64, split-K chunk over V); full M=512 via 16 waves.
// W_v2e streamed from HBM exactly once (disjoint h-rows x v-chunks).
__global__ __launch_bounds__(512)
void imdl_gemm2(const _Float16* __restrict__ P,   // [512, V] f16
                const float* __restrict__ W2,     // [896, V] fp32
                float* __restrict__ acc_out) {    // [512, 896] fp32
    __shared__ _Float16 Bs[64 * BP];
    const int tid   = threadIdx.x;
    const int wave  = tid >> 5;
    const int lane  = tid & 31;
    const int lo    = lane & 15;
    const int hi    = lane >> 4;
    const int hbase = blockIdx.x * 64;
    const int mbase = wave * 32;
    const int s0 = blockIdx.y * SPW;
    const int s1 = (s0 + SPW < KST2) ? (s0 + SPW) : KST2;

    v8f acc[2][4] = {};

    for (int s = s0; s < s1; ++s) {
        const int kc = s * 32;
        {
            const int row = tid >> 3;
            const int col = (tid & 7) * 4;
            const float4 w4 = *(const float4*)(W2 + (size_t)(hbase + row) * VDIM + kc + col);
            v4h h4;
            h4[0] = (_Float16)w4.x; h4[1] = (_Float16)w4.y;
            h4[2] = (_Float16)w4.z; h4[3] = (_Float16)w4.w;
            *(v4h*)(&Bs[row * BP + col]) = h4;
        }
        __syncthreads();

        v16h a[2];
#pragma unroll
        for (int mt = 0; mt < 2; ++mt) {
            const _Float16* ar = P + (size_t)(mbase + mt * 16 + lo) * VDIM + kc;
            v8h a0 = *(const v8h*)(ar + hi * 8);
            v8h a1 = *(const v8h*)(ar + 16 + hi * 8);
            a[mt] = __builtin_shufflevector(a0, a1, 0,1,2,3,4,5,6,7,8,9,10,11,12,13,14,15);
        }
#pragma unroll
        for (int nt = 0; nt < 4; ++nt) {
            const _Float16* br = &Bs[(nt * 16 + lo) * BP + hi * 8];
            v8h b0 = *(const v8h*)br;
            v8h b1 = *(const v8h*)(br + 16);
            v16h b = __builtin_shufflevector(b0, b1, 0,1,2,3,4,5,6,7,8,9,10,11,12,13,14,15);
#pragma unroll
            for (int mt = 0; mt < 2; ++mt)
                acc[mt][nt] = __builtin_amdgcn_wmma_f32_16x16x32_f16(
                    false, a[mt], false, b, (short)0, acc[mt][nt], false, false);
        }
        __syncthreads();
    }

#pragma unroll
    for (int mt = 0; mt < 2; ++mt) {
#pragma unroll
        for (int r = 0; r < 8; ++r) {
            const int m = mbase + mt * 16 + hi * 8 + r;
#pragma unroll
            for (int nt = 0; nt < 4; ++nt)
                atomicAdd(&acc_out[(size_t)m * HDIM + hbase + nt * 16 + lo], acc[mt][nt][r]);
        }
    }
}

__global__ void imdl_final(const float* __restrict__ acc,
                           const float* __restrict__ bias,
                           const float* __restrict__ strength,
                           const float* __restrict__ hidden,
                           float* __restrict__ out) {
    int i = blockIdx.x * blockDim.x + threadIdx.x;
    if (i >= MDIM * HDIM) return;
    out[i] = acc[i] * strength[0] + bias[i % HDIM] + hidden[i];
}

extern "C" void kernel_launch(void* const* d_in, const int* in_sizes, int n_in,
                              void* d_out, int out_size, void* d_ws, size_t ws_size,
                              hipStream_t stream) {
    const float* hidden   = (const float*)d_in[0];   // [2,256,896]
    const float* W_e2v    = (const float*)d_in[1];   // [V,896]
    const float* W_v2e    = (const float*)d_in[2];   // [896,V]
    const float* b_v2e    = (const float*)d_in[3];   // [896]
    const float* strength = (const float*)d_in[4];   // [1]
    const int*   mask_idx = (const int*)d_in[5];     // [2,256,8]
    const int*   mask_len = (const int*)d_in[6];     // [2,256]
    float* out = (float*)d_out;
    (void)in_sizes; (void)n_in; (void)out_size; (void)ws_size;

    char* ws = (char*)d_ws;
    _Float16* A16   = (_Float16*)(ws + 0);          //   917504 B
    unsigned* key   = (unsigned*)(ws + 917504);     //     2048 B
    float*    accum = (float*)   (ws + 919552);     //  1835008 B
    _Float16* P     = (_Float16*)(ws + 2754560);    // 155582464 B (~151 MiB total)

    imdl_prep<<<(MDIM * HDIM + 255) / 256, 256, 0, stream>>>(hidden, A16, accum, key);
    imdl_gemm1<<<VDIM / 64, 512, 0, stream>>>(A16, W_e2v, P, key);
    imdl_softneg<<<(MDIM * (VDIM / 8) + 255) / 256, 256, 0, stream>>>(P, key, mask_len);
    imdl_scatter<<<(MDIM * KTOK + 255) / 256, 256, 0, stream>>>(P, mask_idx, mask_len);
    imdl_gemm2<<<dim3(HDIM / 64, G2Y), 512, 0, stream>>>(P, W_v2e, accum);
    imdl_final<<<(MDIM * HDIM + 255) / 256, 256, 0, stream>>>(accum, b_v2e, strength, hidden, out);
}
